// TemporalAttention_35021163332114
// MI455X (gfx1250) — matrix operand order repaired and checked by reading, not compiled, not run on hardware
//
#include <hip/hip_runtime.h>
#include <hip/hip_bf16.h>
#include <math.h>

// Problem constants (from reference)
#define Hc 128
#define TDc 32
#define Bc 16
#define Nc 256
#define Tc 128

typedef __attribute__((ext_vector_type(16))) __bf16 v16bf;
typedef __attribute__((ext_vector_type(8)))  float  v8f;
typedef __attribute__((ext_vector_type(4)))  unsigned int v4u;
typedef __attribute__((ext_vector_type(8)))  int v8i;
typedef __attribute__((ext_vector_type(4)))  int v4i;

union Frag { v16bf v; v4u u[2]; };

// A-matrix fragment (16x32 bf16), row-major source with leading dim ld (elems).
// ISA 7.12.2: lanes 0-15 row M=lane, VGPR0-3 = K 0..7, VGPR4-7 = K 16..23;
// lanes 16-31 same rows, K 8..15 and K 24..31.
__device__ inline v16bf load_fragA(const __bf16* base, int ld, int row0, int k0) {
    int lane = threadIdx.x & 31;
    const __bf16* p = base + (row0 + (lane & 15)) * ld + k0 + ((lane >> 4) << 3);
    Frag f;
    f.u[0] = *(const v4u*)(p);
    f.u[1] = *(const v4u*)(p + 16);
    return f.v;
}

// B-matrix fragment (32x16 bf16) from COLUMN-major storage, i.e. Bt[n][k] row-major.
__device__ inline v16bf load_fragB(const __bf16* base, int ld, int col0, int k0) {
    int lane = threadIdx.x & 31;
    const __bf16* p = base + (col0 + (lane & 15)) * ld + k0 + ((lane >> 4) << 3);
    Frag f;
    f.u[0] = *(const v4u*)(p);
    f.u[1] = *(const v4u*)(p + 16);
    return f.v;
}

__device__ inline v8f wmma_bf16(v16bf a, v16bf b, v8f c) {
    return __builtin_amdgcn_wmma_f32_16x16x32_bf16(false, a, false, b, (short)0, c,
                                                   false, false);
}

// Tensor Data Mover: DMA a 2D tile (dim1 rows x dim0 elems) global -> LDS.
// size_code: 0=1B,1=2B,2=4B,3=8B.  stride0 in elements.  (ISA ch.8 D# layout.)
__device__ inline void tdm_load_2d(unsigned lds_addr, const void* gptr,
                                   unsigned size_code, unsigned dim0,
                                   unsigned dim1, unsigned stride0) {
    unsigned long long ga = (unsigned long long)(uintptr_t)gptr;
    v4u g0;
    g0[0] = 1u;                                       // count=1, user mode
    g0[1] = lds_addr;                                 // LDS byte address
    g0[2] = (unsigned)(ga & 0xFFFFFFFFu);             // global_addr[31:0]
    g0[3] = (unsigned)((ga >> 32) & 0x1FFFFFFu)       // global_addr[56:32]
          | (2u << 30);                               // type = 2 ("image")
    v8i g1;
    g1[0] = (int)(size_code << 16);                   // wg_mask=0, data_size
    g1[1] = (int)(dim0 << 16);                        // tensor_dim0[15:0] @bit48
    g1[2] = (int)((dim0 >> 16) | (dim1 << 16));       // dim0[31:16], dim1[15:0]
    g1[3] = (int)((dim1 >> 16) | (dim0 << 16));       // dim1[31:16], tile_dim0
    g1[4] = (int)(dim1 & 0xFFFFu);                    // tile_dim1 (tile_dim2=0)
    g1[5] = (int)stride0;                             // tensor_dim0_stride[31:0]
    g1[6] = 0;                                        // stride0 hi, stride1 lo
    g1[7] = 0;                                        // stride1 hi
    v4i z4 = {0, 0, 0, 0};
    v8i z8 = {0, 0, 0, 0, 0, 0, 0, 0};
    __builtin_amdgcn_tensor_load_to_lds(g0, g1, z4, z4, z8, 0);
}

// ---------------------------------------------------------------------------
// P1: bias_bt[b,t,h] = pe[t]@W_in[0:128] + (ex[b,t]@W_t + b_t)@W_in[128:256] + b_in
__global__ void prep_bias_bt(const float* __restrict__ ex, const float* __restrict__ W_t,
                             const float* __restrict__ b_t, const float* __restrict__ W_in,
                             const float* __restrict__ b_in, float* __restrict__ bias_bt) {
    int bt = blockIdx.x;          // b*T + t
    int t  = bt & (Tc - 1);
    int h  = threadIdx.x;
    __shared__ float te[Hc];
    __shared__ float pe[Hc];
    float acc = b_t[h];
    const float* exr = ex + (size_t)bt * TDc;
    #pragma unroll 4
    for (int d = 0; d < TDc; ++d) acc += exr[d] * W_t[d * Hc + h];
    te[h] = acc;
    {
        float div = expf((float)(h & ~1) * (-9.210340371976184f / (float)Hc));
        float ang = (float)t * div;
        pe[h] = (h & 1) ? cosf(ang) : sinf(ang);
    }
    __syncthreads();
    float s = b_in[h];
    #pragma unroll 4
    for (int j = 0; j < Hc; ++j)
        s += pe[j] * W_in[j * Hc + h] + te[j] * W_in[(Hc + j) * Hc + h];
    bias_bt[(size_t)bt * Hc + h] = s;
}

// P2: bias_n[n,h] = E_node[nf[n]] @ W_in[256:384]
__global__ void prep_bias_n(const int* __restrict__ nf, const float* __restrict__ E_node,
                            const float* __restrict__ W_in, float* __restrict__ bias_n) {
    int n = blockIdx.x;
    int h = threadIdx.x;
    const float* e = E_node + (size_t)nf[n] * Hc;
    float s = 0.0f;
    #pragma unroll 4
    for (int j = 0; j < Hc; ++j) s += e[j] * W_in[(2 * Hc + j) * Hc + h];
    bias_n[(size_t)n * Hc + h] = s;
}

// P3: transpose + convert weights to bf16 "Bt" layout: dst[m][n][k] = W_m[k][n]
__global__ void prep_weights(const float* __restrict__ W_in, const float* __restrict__ Wq,
                             const float* __restrict__ Wk, const float* __restrict__ Wv,
                             const float* __restrict__ Wo, __bf16* __restrict__ dst) {
    int m  = blockIdx.x >> 7;
    int nn = blockIdx.x & 127;
    int k  = threadIdx.x;
    const float* src = (m == 0) ? W_in : (m == 1) ? Wq : (m == 2) ? Wk
                      : (m == 3) ? Wv : Wo;
    dst[(size_t)m * Hc * Hc + nn * Hc + k] = (__bf16)src[k * Hc + nn];
}

// ---------------------------------------------------------------------------
// Fused per-(b,n) tile kernel.  256 threads = 8 waves; wave w owns row-block w.
__global__ __launch_bounds__(256) void fused_attn(
    const float* __restrict__ xf, const unsigned char* __restrict__ mask,
    const float* __restrict__ bias_bt, const float* __restrict__ bias_n,
    const __bf16* __restrict__ Wts, const float* __restrict__ b_q,
    const float* __restrict__ b_k, const float* __restrict__ b_v,
    const float* __restrict__ b_o, float* __restrict__ out_rec,
    float* __restrict__ out_conf) {
    // LDS carve (224KB + tail):
    //  buf0..buf3: 128x128 bf16 (32KB each)   bufS: 64KB   buf4: 32KB
    __shared__ __align__(16) char smem[224 * 1024 + 1024];
    __bf16* buf0 = (__bf16*)(smem);               // xf(bf16) -> q -> rec
    __bf16* buf1 = (__bf16*)(smem + 32 * 1024);   // x -> A(bf16)
    __bf16* buf2 = (__bf16*)(smem + 64 * 1024);   // W1t (TDM) -> k
    __bf16* buf3 = (__bf16*)(smem + 96 * 1024);   // v^T
    float*  bufS = (float*)(smem + 128 * 1024);   // bias(TDM) -> Wq/Wk(TDM) -> scores -> Wo(TDM)
    __bf16* WqL  = (__bf16*)(smem + 128 * 1024);
    __bf16* WkL  = (__bf16*)(smem + 128 * 1024 + 32768);
    __bf16* WoL  = (__bf16*)(smem + 128 * 1024);
    __bf16* buf4 = (__bf16*)(smem + 192 * 1024);  // Wv^T (TDM)
    float*  sTm  = (float*)(smem + 224 * 1024);   // t_masking [T]

    const int tile = blockIdx.x;        // b*N + n
    const int b    = tile >> 8;
    const int n    = tile & (Nc - 1);
    const int tid  = threadIdx.x;
    const int lane = tid & 31;
    const int w    = tid >> 5;
    const int colLane = lane & 15;
    const int rowLane = (lane >> 4) << 3;

    // ---- Phase 0: TDM-stage bias/W1t/Wvt, convert xf -> bf16, t_masking -----
    if (w == 0) {
        tdm_load_2d((unsigned)(uintptr_t)bufS, bias_bt + (size_t)b * Tc * Hc,
                    2u, Hc, Tc, Hc);                           // 64KB f32 bias tile
        tdm_load_2d((unsigned)(uintptr_t)buf2, Wts,            // W1t bf16
                    1u, Hc, Hc, Hc);
        tdm_load_2d((unsigned)(uintptr_t)buf4, Wts + 3 * Hc * Hc,  // Wvt bf16
                    1u, Hc, Hc, Hc);
    }
    const float* xsrc = xf + (size_t)tile * (Tc * Hc);
    for (int i = tid; i < (Tc * Hc) / 4; i += 256) {
        float4 v = ((const float4*)xsrc)[i];
        buf0[i * 4 + 0] = (__bf16)v.x;
        buf0[i * 4 + 1] = (__bf16)v.y;
        buf0[i * 4 + 2] = (__bf16)v.z;
        buf0[i * 4 + 3] = (__bf16)v.w;
    }
    if (tid < Tc) {
        const unsigned char* mrow = mask + (size_t)tile * Tc * Hc + (size_t)tid * Hc;
        int allb = 1;
        for (int hh = 0; hh < Hc; ++hh) allb &= (mrow[hh] != 0);
        sTm[tid] = allb ? 1.0f : 0.0f;
    }
    if (w == 0) __builtin_amdgcn_s_wait_tensorcnt(0);
    __syncthreads();

    // ---- Phase 1: x = GELU(xf @ W1 + bias_bt + bias_n) ----------------------
    {
        v16bf a0 = load_fragA(buf0, Hc, w * 16, 0);
        v16bf a1 = load_fragA(buf0, Hc, w * 16, 32);
        v16bf a2 = load_fragA(buf0, Hc, w * 16, 64);
        v16bf a3 = load_fragA(buf0, Hc, w * 16, 96);
        for (int nb = 0; nb < 8; ++nb) {
            v8f acc = {};
            acc = wmma_bf16(a0, load_fragB(buf2, Hc, nb * 16, 0),  acc);
            acc = wmma_bf16(a1, load_fragB(buf2, Hc, nb * 16, 32), acc);
            acc = wmma_bf16(a2, load_fragB(buf2, Hc, nb * 16, 64), acc);
            acc = wmma_bf16(a3, load_fragB(buf2, Hc, nb * 16, 96), acc);
            int col = nb * 16 + colLane;
            float bn = bias_n[(size_t)n * Hc + col];
            #pragma unroll
            for (int r = 0; r < 8; ++r) {
                int row = w * 16 + rowLane + r;
                float pre = acc[r] + bufS[row * Hc + col] + bn;
                float g = 0.5f * pre * (1.0f + erff(pre * 0.70710678118654752f));
                buf1[row * Hc + col] = (__bf16)g;
            }
        }
    }
    __syncthreads();

    // TDM-stage Wq/Wk over the (now dead) bias tile
    if (w == 0) {
        tdm_load_2d((unsigned)(uintptr_t)WqL, Wts + 1 * Hc * Hc, 1u, Hc, Hc, Hc);
        tdm_load_2d((unsigned)(uintptr_t)WkL, Wts + 2 * Hc * Hc, 1u, Hc, Hc, Hc);
        __builtin_amdgcn_s_wait_tensorcnt(0);
    }
    __syncthreads();

    // ---- Phase 2: q,k,v projections (q->buf0, k->buf2, v^T->buf3) -----------
    {
        v16bf a0 = load_fragA(buf1, Hc, w * 16, 0);
        v16bf a1 = load_fragA(buf1, Hc, w * 16, 32);
        v16bf a2 = load_fragA(buf1, Hc, w * 16, 64);
        v16bf a3 = load_fragA(buf1, Hc, w * 16, 96);
        for (int nb = 0; nb < 8; ++nb) {
            int col = nb * 16 + colLane;
            // q
            v8f acc = {};
            acc = wmma_bf16(a0, load_fragB(WqL, Hc, nb * 16, 0),  acc);
            acc = wmma_bf16(a1, load_fragB(WqL, Hc, nb * 16, 32), acc);
            acc = wmma_bf16(a2, load_fragB(WqL, Hc, nb * 16, 64), acc);
            acc = wmma_bf16(a3, load_fragB(WqL, Hc, nb * 16, 96), acc);
            float bq = b_q[col];
            #pragma unroll
            for (int r = 0; r < 8; ++r) {
                int row = w * 16 + rowLane + r;
                buf0[row * Hc + col] = (__bf16)(acc[r] + bq);
            }
            // k
            v8f acck = {};
            acck = wmma_bf16(a0, load_fragB(WkL, Hc, nb * 16, 0),  acck);
            acck = wmma_bf16(a1, load_fragB(WkL, Hc, nb * 16, 32), acck);
            acck = wmma_bf16(a2, load_fragB(WkL, Hc, nb * 16, 64), acck);
            acck = wmma_bf16(a3, load_fragB(WkL, Hc, nb * 16, 96), acck);
            float bk = b_k[col];
            #pragma unroll
            for (int r = 0; r < 8; ++r) {
                int row = w * 16 + rowLane + r;
                buf2[row * Hc + col] = (__bf16)(acck[r] + bk);
            }
            // v (store transposed: buf3[h][t])
            v8f accv = {};
            accv = wmma_bf16(a0, load_fragB(buf4, Hc, nb * 16, 0),  accv);
            accv = wmma_bf16(a1, load_fragB(buf4, Hc, nb * 16, 32), accv);
            accv = wmma_bf16(a2, load_fragB(buf4, Hc, nb * 16, 64), accv);
            accv = wmma_bf16(a3, load_fragB(buf4, Hc, nb * 16, 96), accv);
            float bv = b_v[col];
            #pragma unroll
            for (int r = 0; r < 8; ++r) {
                int row = w * 16 + rowLane + r;
                buf3[col * Tc + row] = (__bf16)(accv[r] + bv);
            }
        }
    }
    __syncthreads();

    // ---- Phase 3: scores = q @ k^T * rsqrt(H)  (f32 to bufS) ----------------
    {
        v16bf a0 = load_fragA(buf0, Hc, w * 16, 0);
        v16bf a1 = load_fragA(buf0, Hc, w * 16, 32);
        v16bf a2 = load_fragA(buf0, Hc, w * 16, 64);
        v16bf a3 = load_fragA(buf0, Hc, w * 16, 96);
        const float scale = 0.08838834764831845f;   // 1/sqrt(128)
        for (int nb = 0; nb < 8; ++nb) {
            v8f acc = {};
            acc = wmma_bf16(a0, load_fragB(buf2, Hc, nb * 16, 0),  acc);
            acc = wmma_bf16(a1, load_fragB(buf2, Hc, nb * 16, 32), acc);
            acc = wmma_bf16(a2, load_fragB(buf2, Hc, nb * 16, 64), acc);
            acc = wmma_bf16(a3, load_fragB(buf2, Hc, nb * 16, 96), acc);
            int col = nb * 16 + colLane;
            #pragma unroll
            for (int r = 0; r < 8; ++r) {
                int row = w * 16 + rowLane + r;
                bufS[row * Tc + col] = acc[r] * scale;
            }
        }
    }
    __syncthreads();

    // ---- Phase 4: softmax + confidence; A (bf16) -> buf1 --------------------
    if (tid < Tc) {
        int s = tid;
        float* srow = bufS + s * Tc;
        float mx = -3.4e38f;
        for (int t = 0; t < Tc; ++t) mx = fmaxf(mx, srow[t]);
        float sum = 0.0f;
        for (int t = 0; t < Tc; ++t) { float e = expf(srow[t] - mx); srow[t] = e; sum += e; }
        float inv = 1.0f / sum;
        float conf = 0.0f;
        for (int t = 0; t < Tc; ++t) {
            float a = srow[t] * inv;
            buf1[s * Tc + t] = (__bf16)a;
            conf += a * sTm[t];
        }
        out_conf[(size_t)tile * Tc + s] = 1.0f - conf;
    }
    __syncthreads();

    // ---- Phase 5: rec = A @ v ; TDM-stage Wo concurrently -------------------
    if (w == 0) {
        tdm_load_2d((unsigned)(uintptr_t)WoL, Wts + 4 * Hc * Hc, 1u, Hc, Hc, Hc);
    }
    {
        v16bf a0 = load_fragA(buf1, Tc, w * 16, 0);
        v16bf a1 = load_fragA(buf1, Tc, w * 16, 32);
        v16bf a2 = load_fragA(buf1, Tc, w * 16, 64);
        v16bf a3 = load_fragA(buf1, Tc, w * 16, 96);
        for (int nb = 0; nb < 8; ++nb) {
            v8f acc = {};
            acc = wmma_bf16(a0, load_fragB(buf3, Tc, nb * 16, 0),  acc);
            acc = wmma_bf16(a1, load_fragB(buf3, Tc, nb * 16, 32), acc);
            acc = wmma_bf16(a2, load_fragB(buf3, Tc, nb * 16, 64), acc);
            acc = wmma_bf16(a3, load_fragB(buf3, Tc, nb * 16, 96), acc);
            int col = nb * 16 + colLane;
            #pragma unroll
            for (int r = 0; r < 8; ++r) {
                int row = w * 16 + rowLane + r;
                buf0[row * Hc + col] = (__bf16)acc[r];
            }
        }
    }
    if (w == 0) __builtin_amdgcn_s_wait_tensorcnt(0);
    __syncthreads();

    // ---- Phase 6: out = rec @ W_o + b_o  (f32 to global) --------------------
    {
        v16bf a0 = load_fragA(buf0, Hc, w * 16, 0);
        v16bf a1 = load_fragA(buf0, Hc, w * 16, 32);
        v16bf a2 = load_fragA(buf0, Hc, w * 16, 64);
        v16bf a3 = load_fragA(buf0, Hc, w * 16, 96);
        for (int nb = 0; nb < 8; ++nb) {
            v8f acc = {};
            acc = wmma_bf16(a0, load_fragB(WoL, Hc, nb * 16, 0),  acc);
            acc = wmma_bf16(a1, load_fragB(WoL, Hc, nb * 16, 32), acc);
            acc = wmma_bf16(a2, load_fragB(WoL, Hc, nb * 16, 64), acc);
            acc = wmma_bf16(a3, load_fragB(WoL, Hc, nb * 16, 96), acc);
            int col = nb * 16 + colLane;
            float bo = b_o[col];
            #pragma unroll
            for (int r = 0; r < 8; ++r) {
                int row = w * 16 + rowLane + r;
                out_rec[((size_t)tile * Tc + row) * Hc + col] = acc[r] + bo;
            }
        }
    }
}

// ---------------------------------------------------------------------------
extern "C" void kernel_launch(void* const* d_in, const int* in_sizes, int n_in,
                              void* d_out, int out_size, void* d_ws, size_t ws_size,
                              hipStream_t stream) {
    const float* xf     = (const float*)d_in[0];
    const float* ex     = (const float*)d_in[1];
    const int*   nf     = (const int*)d_in[2];
    const unsigned char* mask = (const unsigned char*)d_in[3];
    const float* W_in   = (const float*)d_in[4];
    const float* b_in   = (const float*)d_in[5];
    const float* W_t    = (const float*)d_in[6];
    const float* b_t    = (const float*)d_in[7];
    const float* E_node = (const float*)d_in[8];
    const float* W_q    = (const float*)d_in[9];
    const float* b_q    = (const float*)d_in[10];
    const float* W_k    = (const float*)d_in[11];
    const float* b_k    = (const float*)d_in[12];
    const float* W_v    = (const float*)d_in[13];
    const float* b_v    = (const float*)d_in[14];
    const float* W_o    = (const float*)d_in[15];
    const float* b_o    = (const float*)d_in[16];

    char* ws = (char*)d_ws;
    float*  bias_bt = (float*)ws;                                    // B*T*H f32 = 1 MB
    float*  bias_n  = (float*)(ws + (size_t)(1 << 20));              // N*H f32  = 128 KB
    __bf16* Wts     = (__bf16*)(ws + (size_t)(1 << 20) + (1 << 17)); // 5*128*128 bf16

    prep_bias_bt<<<Bc * Tc, Hc, 0, stream>>>(ex, W_t, b_t, W_in, b_in, bias_bt);
    prep_bias_n<<<Nc, Hc, 0, stream>>>(nf, E_node, W_in, bias_n);
    prep_weights<<<5 * Hc, Hc, 0, stream>>>(W_in, W_q, W_k, W_v, W_o, Wts);

    float* out_rec  = (float*)d_out;
    float* out_conf = out_rec + (size_t)Bc * Nc * Tc * Hc;
    fused_attn<<<Bc * Nc, 256, 0, stream>>>(xf, mask, bias_bt, bias_n, Wts,
                                            b_q, b_k, b_v, b_o, out_rec, out_conf);
}
